// MultiLinearAttention_40570261078615
// MI455X (gfx1250) — matrix operand order repaired and checked by reading, not compiled, not run on hardware
//
#include <hip/hip_runtime.h>
#include <hip/hip_bf16.h>
#include <stdint.h>

typedef __attribute__((ext_vector_type(16))) __bf16 v16bf;
typedef __attribute__((ext_vector_type(8)))  __bf16 v8bf;
typedef __attribute__((ext_vector_type(8)))  float  v8f;

#define LDP 72   // padded LDS row stride (elements); 72*2B=144B keeps 16B alignment

// ---- CDNA5 async global->LDS copy (ASYNCcnt-tracked) -----------------------
__device__ __forceinline__ void async_ld_b128(uint32_t lds_off, const void* gptr) {
  asm volatile("global_load_async_to_lds_b128 %0, %1, off"
               :: "v"(lds_off), "v"(gptr) : "memory");
}
__device__ __forceinline__ void wait_async0() {
  asm volatile("s_wait_asynccnt 0x0" ::: "memory");
}

// ---- WMMA fragment helpers (wave32, 16x16x32 bf16) -------------------------
// A-matrix (16x32, 16-bit): lane L (m = mo + L%16, half = L/16):
//   VGPR0..3 hold K = half*8 + {0..7}, VGPR4..7 hold K = 16 + half*8 + {0..7}
// => two contiguous 8-element (16B) runs per lane.
__device__ __forceinline__ v16bf load_a_frag(const __bf16* base, int ld,
                                             int mo, int ko, int lane) {
  int half = lane >> 4;
  int m    = mo + (lane & 15);
  const __bf16* r = base + m * ld;
  union { v16bf v; v8bf h[2]; } u;
  u.h[0] = *(const v8bf*)(r + ko + half * 8);
  u.h[1] = *(const v8bf*)(r + ko + 16 + half * 8);
  return u.v;
}

// B-matrix (32x16) loaded from *transposed* [N][K] LDS storage: lane L holds
// column n = L%16, K = ko + (L/16)*8 + {0..7} and ko + 16 + (L/16)*8 + {0..7}.
__device__ __forceinline__ v16bf load_bt_frag(const __bf16* bt, int ld,
                                              int no, int ko, int lane) {
  int half = lane >> 4;
  int n    = no + (lane & 15);
  const __bf16* r = bt + n * ld + ko + half * 8;
  union { v16bf v; v8bf h[2]; } u;
  u.h[0] = *(const v8bf*)(r);
  u.h[1] = *(const v8bf*)(r + 16);
  return u.v;
}

// Two-tile matmul: {c0,c1} += A[row tile mo] * B[:, no0 / no1], K = 64.
// A fragments are loaded ONCE and feed two back-to-back WMMAs (SRC0 reuse).
__device__ __forceinline__ void mm64x2(const __bf16* A, int lda, int mo,
                                       const __bf16* BT, int ldb, int no0, int no1,
                                       v8f& c0, v8f& c1, int lane) {
  v16bf a0  = load_a_frag(A, lda, mo, 0, lane);
  v16bf b00 = load_bt_frag(BT, ldb, no0, 0, lane);
  v16bf b10 = load_bt_frag(BT, ldb, no1, 0, lane);
  c0 = __builtin_amdgcn_wmma_f32_16x16x32_bf16(false, a0, false, b00,
                                               (short)0, c0, false, false);
  c1 = __builtin_amdgcn_wmma_f32_16x16x32_bf16(false, a0, false, b10,
                                               (short)0, c1, false, false);
  v16bf a1  = load_a_frag(A, lda, mo, 32, lane);
  v16bf b01 = load_bt_frag(BT, ldb, no0, 32, lane);
  v16bf b11 = load_bt_frag(BT, ldb, no1, 32, lane);
  c0 = __builtin_amdgcn_wmma_f32_16x16x32_bf16(false, a1, false, b01,
                                               (short)0, c0, false, false);
  c1 = __builtin_amdgcn_wmma_f32_16x16x32_bf16(false, a1, false, b11,
                                               (short)0, c1, false, false);
}

// ---------------------------------------------------------------------------
// Kernel 1: chunked causal linear attention. One WG per (b,h); 8 waves.
// Async-pipelined X staging: chunk ch+1 DMAs into LDS while ch computes.
// ---------------------------------------------------------------------------
__global__ __launch_bounds__(256) void k_linattn(
    const float* __restrict__ x,  const float* __restrict__ Wq,
    const float* __restrict__ Wk, const float* __restrict__ Wv,
    __bf16* __restrict__ Y)
{
  constexpr int S = 4096, D = 1024, H = 16, HD = 64, O = 64, C = 64;
  constexpr int NC = S / C;

  __shared__ __attribute__((aligned(16))) __bf16 Xc [64 * LDP];
  __shared__ __attribute__((aligned(16))) __bf16 WqT[64 * LDP];
  __shared__ __attribute__((aligned(16))) __bf16 WkT[64 * LDP];
  __shared__ __attribute__((aligned(16))) __bf16 WvT[64 * LDP];
  __shared__ __attribute__((aligned(16))) __bf16 Qs [64 * LDP]; // [t][f]
  __shared__ __attribute__((aligned(16))) __bf16 Ks [64 * LDP]; // [t][f]
  __shared__ __attribute__((aligned(16))) __bf16 KT [64 * LDP]; // [f][t]
  __shared__ __attribute__((aligned(16))) __bf16 Vs [64 * LDP]; // [o][t]
  __shared__ __attribute__((aligned(16))) __bf16 As [64 * LDP]; // masked QK^T
  __shared__ __attribute__((aligned(16))) __bf16 SkvT[64 * LDP];// [o][e]
  __shared__ float Skv[64 * LDP];                               // f32 state [e][o]
  __shared__ __attribute__((aligned(16))) float Xf[2][64 * 64]; // async staging
  __shared__ float zs[64];
  __shared__ float dn[64];

  const int wg = blockIdx.x;            // b*H + h
  const int b = wg / H, h = wg % H;
  const int tid = threadIdx.x;
  const int wave = tid >> 5, lane = tid & 31;
  const int n = lane & 15, mb = (lane >> 4) * 8;

  // Stage per-head weights transposed ([o][d]) as bf16; zero state.
  for (int i = tid; i < 64 * 64; i += 256) {
    int d = i >> 6, o = i & 63;
    WqT[o * LDP + d] = (__bf16)Wq[(h * HD + d) * O + o];
    WkT[o * LDP + d] = (__bf16)Wk[(h * HD + d) * O + o];
    WvT[o * LDP + d] = (__bf16)Wv[(h * HD + d) * O + o];
  }
  for (int i = tid; i < 64 * LDP; i += 256) Skv[i] = 0.f;
  if (tid < 64) zs[tid] = 0.f;

  // Prologue: async-stage chunk 0 (4 x b128 per thread = 16KB per chunk).
  #pragma unroll
  for (int k = 0; k < 4; ++k) {
    int s = tid + k * 256;               // float4 slot
    int t = s >> 4, cc = (s & 15) * 4;
    async_ld_b128((uint32_t)(uintptr_t)&Xf[0][t * 64 + cc],
                  x + ((size_t)(b * S + t)) * D + h * HD + cc);
  }
  __syncthreads();

  int buf = 0;
  for (int ch = 0; ch < NC; ++ch) {
    const int t0 = ch * C;

    // 1) drain own async ops, convert own slots f32 -> bf16 (per-wave
    //    ASYNCcnt ordering covers the wait->convert dependency).
    wait_async0();
    #pragma unroll
    for (int k = 0; k < 4; ++k) {
      int s = tid + k * 256;
      int t = s >> 4, cc = (s & 15) * 4;
      const float4 f = *(const float4*)&Xf[buf][t * 64 + cc];
      __bf16* dst = &Xc[t * LDP + cc];
      dst[0] = (__bf16)f.x; dst[1] = (__bf16)f.y;
      dst[2] = (__bf16)f.z; dst[3] = (__bf16)f.w;
    }
    __syncthreads();

    // 1b) kick off async stage of next chunk; overlaps with WMMA work below.
    if (ch + 1 < NC) {
      #pragma unroll
      for (int k = 0; k < 4; ++k) {
        int s = tid + k * 256;
        int t = s >> 4, cc = (s & 15) * 4;
        async_ld_b128((uint32_t)(uintptr_t)&Xf[buf ^ 1][t * 64 + cc],
                      x + ((size_t)(b * S + t0 + C + t)) * D + h * HD + cc);
      }
    }

    // 2) projections: 24 pair-jobs (A fragment shared within a pair).
    for (int p = wave * 3; p < wave * 3 + 3; ++p) {
      int j = 2 * p;                      // even -> pair never straddles 'which'
      int which = j / 16, ti = j % 16;    // ti even
      int mo = (ti >> 2) * 16;
      int no0 = (ti & 3) * 16, no1 = no0 + 16;
      const __bf16* WT = (which == 0) ? WqT : (which == 1) ? WkT : WvT;
      v8f c0 = {}, c1 = {};
      mm64x2(Xc, LDP, mo, WT, LDP, no0, no1, c0, c1, lane);
      if (which == 2) {                    // V stored transposed [o][t]
        #pragma unroll
        for (int r = 0; r < 8; ++r) {
          Vs[(no0 + n) * LDP + (mo + mb + r)] = (__bf16)c0[r];
          Vs[(no1 + n) * LDP + (mo + mb + r)] = (__bf16)c1[r];
        }
      } else {
        #pragma unroll
        for (int r = 0; r < 8; ++r) {
          float u0 = c0[r], u1 = c1[r];
          __bf16 p0 = (__bf16)((u0 > 0.f) ? (u0 + 1.f) : __expf(u0)); // elu+1
          __bf16 p1 = (__bf16)((u1 > 0.f) ? (u1 + 1.f) : __expf(u1));
          int row = (mo + mb + r) * LDP;
          if (which == 0) { Qs[row + no0 + n] = p0; Qs[row + no1 + n] = p1; }
          else {
            Ks[row + no0 + n] = p0; Ks[row + no1 + n] = p1;
            KT[(no0 + n) * LDP + (mo + mb + r)] = p0;
            KT[(no1 + n) * LDP + (mo + mb + r)] = p1;
          }
        }
      }
    }
    __syncthreads();

    // 3) A = Q K^T with causal mask (s <= t); two tiles share A rows.
    {
      int ti = wave * 2;
      int mo = (ti >> 2) * 16, no0 = (ti & 3) * 16, no1 = no0 + 16;
      v8f c0 = {}, c1 = {};
      mm64x2(Qs, LDP, mo, Ks, LDP, no0, no1, c0, c1, lane);
      #pragma unroll
      for (int r = 0; r < 8; ++r) {
        int t = mo + mb + r;
        int s0 = no0 + n, s1 = no1 + n;
        As[t * LDP + s0] = (s0 <= t) ? (__bf16)c0[r] : (__bf16)0.f;
        As[t * LDP + s1] = (s1 <= t) ? (__bf16)c1[r] : (__bf16)0.f;
      }
    }
    __syncthreads();

    // 4) denominator + f32 state snapshot -> bf16 transposed
    for (int i = tid; i < 64 * 64; i += 256) {
      int e = i >> 6, o = i & 63;
      SkvT[o * LDP + e] = (__bf16)Skv[e * LDP + o];
    }
    if (tid < 64) {
      int t = tid;
      float rs = 0.f;
      for (int s = 0; s < 64; ++s) rs += (float)As[t * LDP + s];
      float qz = 0.f;
      for (int f = 0; f < 64; ++f) qz += (float)Qs[t * LDP + f] * zs[f];
      dn[t] = 1.f / (rs + qz + 1e-6f);
    }
    __syncthreads();

    // 5) y = (Q*Skv_prev + (A∘M)*V) * dn  -> global Y (bf16)
    {
      int ti = wave * 2;
      int mo = (ti >> 2) * 16, no0 = (ti & 3) * 16, no1 = no0 + 16;
      v8f c0 = {}, c1 = {};
      mm64x2(Qs, LDP, mo, SkvT, LDP, no0, no1, c0, c1, lane);
      mm64x2(As, LDP, mo, Vs,   LDP, no0, no1, c0, c1, lane);
      #pragma unroll
      for (int r = 0; r < 8; ++r) {
        int t = mo + mb + r;
        float s = dn[t];
        size_t base = ((size_t)(b * S + t0 + t)) * (H * O) + h * O;
        Y[base + no0 + n] = (__bf16)(c0[r] * s);
        Y[base + no1 + n] = (__bf16)(c1[r] * s);
      }
    }

    // 6) state update: Skv += K^T V (f32 accumulate), z += colsum(K)
    if (tid < 64) {
      int f = tid;
      float s = 0.f;
      for (int t = 0; t < 64; ++t) s += (float)KT[f * LDP + t];
      zs[f] += s;
    }
    {
      int ti = wave * 2;
      int mo = (ti >> 2) * 16, no0 = (ti & 3) * 16, no1 = no0 + 16;
      v8f c0, c1;
      #pragma unroll
      for (int r = 0; r < 8; ++r) {
        c0[r] = Skv[(mo + mb + r) * LDP + no0 + n];
        c1[r] = Skv[(mo + mb + r) * LDP + no1 + n];
      }
      mm64x2(KT, LDP, mo, Vs, LDP, no0, no1, c0, c1, lane);
      #pragma unroll
      for (int r = 0; r < 8; ++r) {
        Skv[(mo + mb + r) * LDP + no0 + n] = c0[r];
        Skv[(mo + mb + r) * LDP + no1 + n] = c1[r];
      }
    }
    __syncthreads();
    buf ^= 1;
  }
}

// ---------------------------------------------------------------------------
// Kernel 2: out[16384,64] = Y[16384,1024](bf16) @ Wp[1024,64] (f32 out)
// ---------------------------------------------------------------------------
__global__ __launch_bounds__(256) void k_proj(
    const __bf16* __restrict__ Y, const float* __restrict__ Wp,
    float* __restrict__ out)
{
  constexpr int KD = 1024, O = 64;
  __shared__ __attribute__((aligned(16))) __bf16 WpT[64 * LDP]; // [o][k]
  const int row0 = blockIdx.x * 64;
  const int tid = threadIdx.x, wave = tid >> 5, lane = tid & 31;
  const int n = lane & 15, mb = (lane >> 4) * 8;

  const int ti = wave * 2;                 // two tiles share A rows
  const int mo = (ti >> 2) * 16, no0 = (ti & 3) * 16, no1 = no0 + 16;
  v8f c0 = {}, c1 = {};

  for (int kb = 0; kb < KD; kb += 64) {
    // prefetch next Wp block into cache while this block computes
    if (kb + 64 < KD)
      __builtin_prefetch(Wp + (size_t)(kb + 64) * O + tid * 16, 0, 0);

    for (int i = tid; i < 64 * 64; i += 256) {
      int k = i >> 6, o = i & 63;
      WpT[o * LDP + k] = (__bf16)Wp[(size_t)(kb + k) * O + o];
    }
    __syncthreads();

    int half = lane >> 4;
    int m = row0 + mo + (lane & 15);
    const __bf16* rp = Y + (size_t)m * KD + kb;  // A frags straight from global
    union { v16bf v; v8bf h[2]; } a0, a1;
    a0.h[0] = *(const v8bf*)(rp + half * 8);
    a0.h[1] = *(const v8bf*)(rp + 16 + half * 8);
    a1.h[0] = *(const v8bf*)(rp + 32 + half * 8);
    a1.h[1] = *(const v8bf*)(rp + 48 + half * 8);
    v16bf b00 = load_bt_frag(WpT, LDP, no0, 0, lane);
    v16bf b10 = load_bt_frag(WpT, LDP, no1, 0, lane);
    c0 = __builtin_amdgcn_wmma_f32_16x16x32_bf16(false, a0.v, false, b00,
                                                 (short)0, c0, false, false);
    c1 = __builtin_amdgcn_wmma_f32_16x16x32_bf16(false, a0.v, false, b10,
                                                 (short)0, c1, false, false);
    v16bf b01 = load_bt_frag(WpT, LDP, no0, 32, lane);
    v16bf b11 = load_bt_frag(WpT, LDP, no1, 32, lane);
    c0 = __builtin_amdgcn_wmma_f32_16x16x32_bf16(false, a1.v, false, b01,
                                                 (short)0, c0, false, false);
    c1 = __builtin_amdgcn_wmma_f32_16x16x32_bf16(false, a1.v, false, b11,
                                                 (short)0, c1, false, false);
    __syncthreads();
  }

  #pragma unroll
  for (int r = 0; r < 8; ++r) {
    size_t base = (size_t)(row0 + mo + mb + r) * O;
    out[base + no0 + n] = c0[r];
    out[base + no1 + n] = c1[r];
  }
}

// ---------------------------------------------------------------------------
extern "C" void kernel_launch(void* const* d_in, const int* in_sizes, int n_in,
                              void* d_out, int out_size, void* d_ws, size_t ws_size,
                              hipStream_t stream) {
  const float* x  = (const float*)d_in[0];
  const float* Wq = (const float*)d_in[1];
  const float* Wk = (const float*)d_in[2];
  const float* Wv = (const float*)d_in[3];
  const float* Wp = (const float*)d_in[4];
  __bf16* Yws = (__bf16*)d_ws;            // 16384 x 1024 bf16 = 32 MB scratch
  float*  out = (float*)d_out;            // 16384 x 64 f32

  k_linattn<<<dim3(64),  dim3(256), 0, stream>>>(x, Wq, Wk, Wv, Yws);
  k_proj  <<<dim3(256), dim3(256), 0, stream>>>(Yws, Wp, out);

  (void)in_sizes; (void)n_in; (void)out_size; (void)ws_size;
}
